// CrossAttentionHedging_48473000902914
// MI455X (gfx1250) — matrix operand order repaired and checked by reading, not compiled, not run on hardware
//
#include <hip/hip_runtime.h>
#include <hip/hip_bf16.h>
#include <math.h>

#define D_MODEL 512
#define NHEAD   8
#define HD      64
#define BATCH   16
#define SEQ     1024
#define NROWS   (BATCH*SEQ)
#define LN_EPS  1e-5f

typedef __bf16 bf16;
typedef __bf16 bf16x16 __attribute__((ext_vector_type(16)));
typedef float  f32x8   __attribute__((ext_vector_type(8)));
typedef unsigned int u32x4 __attribute__((ext_vector_type(4)));
typedef int          i32x4 __attribute__((ext_vector_type(4)));
typedef int          i32x8 __attribute__((ext_vector_type(8)));

union Frag { bf16x16 v; uint4 u[2]; };
union V8   { uint4 u; bf16 e[8]; };

#if __has_builtin(__builtin_amdgcn_tensor_load_to_lds) && __has_builtin(__builtin_amdgcn_s_wait_tensorcnt)
#define HAVE_TDM 1
#else
#define HAVE_TDM 0
#endif

static __device__ __forceinline__ f32x8 zero8() {
    f32x8 z;
#pragma unroll
    for (int i = 0; i < 8; i++) z[i] = 0.0f;
    return z;
}

static __device__ __forceinline__ f32x8 wmma_bf16(const Frag& a, const Frag& b, f32x8 c) {
    return __builtin_amdgcn_wmma_f32_16x16x32_bf16(false, a.v, false, b.v, (short)0, c, false, false);
}

// ---------------------------------------------------------------------------
// fp32 -> bf16 convert (weights)
// ---------------------------------------------------------------------------
__global__ void k_f32_to_bf16(const float* __restrict__ in, bf16* __restrict__ out, int n) {
    int i = blockIdx.x * blockDim.x + threadIdx.x;
    if (i < n) out[i] = (bf16)in[i];
}

// ---------------------------------------------------------------------------
// input projections: history = feat@hist_w^T+hist_b ; x = feat@query_w^T+query_b
// ---------------------------------------------------------------------------
__global__ void k_input_proj(const float* __restrict__ feat,
                             const float* __restrict__ hw, const float* __restrict__ hb,
                             const float* __restrict__ qw, const float* __restrict__ qb,
                             float* __restrict__ xf, bf16* __restrict__ xb,
                             bf16* __restrict__ histb) {
    int idx = blockIdx.x * blockDim.x + threadIdx.x;  // over NROWS*D_MODEL
    int row = idx >> 9;
    int d   = idx & 511;
    const float* f = feat + (size_t)row * 16;
    float ah = hb[d], aq = qb[d];
#pragma unroll
    for (int i = 0; i < 16; i++) {
        float fv = f[i];
        ah += fv * hw[d * 16 + i];
        aq += fv * qw[d * 16 + i];
    }
    xf[idx]    = aq;
    xb[idx]    = (bf16)aq;
    histb[idx] = (bf16)ah;
}

// ---------------------------------------------------------------------------
// WMMA GEMM: C[M,N] = A[M,K] @ W[N,K]^T + bias[N]
// wave tile = 32(M) x 64(N); double-buffered fragment loads; K step 32.
// ---------------------------------------------------------------------------
static __device__ __forceinline__ void load_afrag(Frag a[2], const bf16* __restrict__ A,
                                                  int K, int m0, int ml, int kh, int k0) {
#pragma unroll
    for (int i = 0; i < 2; i++) {
        const bf16* Ap = A + (size_t)(m0 + i * 16 + ml) * K + k0 + kh * 16;
        a[i].u[0] = *(const uint4*)Ap;          // K = [kh*16, kh*16+8)
        a[i].u[1] = *(const uint4*)(Ap + 8);    // K = [kh*16+8, kh*16+16)
    }
}
static __device__ __forceinline__ void load_bfrag(Frag bf[4], const bf16* __restrict__ W,
                                                  int K, int n0, int ml, int kh, int k0) {
#pragma unroll
    for (int j = 0; j < 4; j++) {
        const bf16* Wp = W + (size_t)(n0 + j * 16 + ml) * K + k0 + kh * 16;
        bf[j].u[0] = *(const uint4*)Wp;         // 16 contiguous K values
        bf[j].u[1] = *(const uint4*)(Wp + 8);
    }
}
static __device__ __forceinline__ void mma8(f32x8 c[2][4], const Frag a[2], const Frag bfr[4]) {
#pragma unroll
    for (int i = 0; i < 2; i++)
#pragma unroll
        for (int j = 0; j < 4; j++)
            c[i][j] = wmma_bf16(a[i], bfr[j], c[i][j]);
}

__global__ __launch_bounds__(256) void k_gemm(const bf16* __restrict__ A,
                                              const bf16* __restrict__ W,
                                              const float* __restrict__ bias,
                                              float* __restrict__ outF,
                                              bf16* __restrict__ outB,
                                              int M, int N, int K) {
    int wid  = (blockIdx.x * 256 + threadIdx.x) >> 5;
    int lane = threadIdx.x & 31;
    int nStr = N >> 6;
    int mt = wid / nStr;
    int nt = wid % nStr;
    if (mt >= (M >> 5)) return;
    int m0 = mt << 5;
    int n0 = nt << 6;
    int ml = lane & 15;   // A row / B column / C column within tile
    int kh = lane >> 4;   // half-wave selector

    f32x8 c[2][4];
#pragma unroll
    for (int j = 0; j < 4; j++) {
        float bv = bias ? bias[n0 + j * 16 + ml] : 0.0f;
#pragma unroll
        for (int i = 0; i < 2; i++)
#pragma unroll
            for (int r = 0; r < 8; r++) c[i][j][r] = bv;
    }

    // ping-pong double buffering over K (K assumed multiple of 64)
    Frag aA[2], bA[4], aB[2], bB[4];
    load_afrag(aA, A, K, m0, ml, kh, 0);
    load_bfrag(bA, W, K, n0, ml, kh, 0);
    for (int k0 = 0; k0 < K; k0 += 64) {
        load_afrag(aB, A, K, m0, ml, kh, k0 + 32);
        load_bfrag(bB, W, K, n0, ml, kh, k0 + 32);
        mma8(c, aA, bA);
        if (k0 + 64 < K) {
            load_afrag(aA, A, K, m0, ml, kh, k0 + 64);
            load_bfrag(bA, W, K, n0, ml, kh, k0 + 64);
        }
        mma8(c, aB, bB);
    }

#pragma unroll
    for (int i = 0; i < 2; i++)
#pragma unroll
        for (int j = 0; j < 4; j++) {
            int n = n0 + j * 16 + ml;
#pragma unroll
            for (int r = 0; r < 8; r++) {
                int m = m0 + i * 16 + kh * 8 + r;
                if (outF) outF[(size_t)m * N + n] = c[i][j][r];
                if (outB) outB[(size_t)m * N + n] = (bf16)c[i][j][r];
            }
        }
}

// ---------------------------------------------------------------------------
// Causal cross-attention, flash-style online softmax.
// Block = (batch, head, 64 query rows); 4 waves, 16 q rows each.
// K tile staged via Tensor Data Mover (with LDS row padding); V transposed
// manually (TDM cannot transpose) overlapping the TDM transfer.
// ---------------------------------------------------------------------------
__global__ __launch_bounds__(128) void k_attn(const bf16* __restrict__ Q,
                                              const bf16* __restrict__ Kt,
                                              const bf16* __restrict__ Vt,
                                              bf16* __restrict__ Ctx) {
    __shared__ bf16 sK[64][72];        // [key][hd]       (padded rows)
    __shared__ bf16 sVT[64][72];       // [hd][key]       (transposed V)
    __shared__ bf16 sP[4][16][72];     // per-wave probs  [qrow][key]

    const int nqt = SEQ / 64;
    int qt = blockIdx.x % nqt;
    int bh = blockIdx.x / nqt;
    int h = bh % NHEAD, b = bh / NHEAD;
    int lane = threadIdx.x & 31;
    int wave = threadIdx.x >> 5;
    int ml = lane & 15, kh = lane >> 4;
    int q0 = qt * 64 + wave * 16;
    const float scale = 0.125f;  // 1/sqrt(64)

#if HAVE_TDM
    // LDS byte offset of sK (generic LDS addresses carry the offset in [31:0])
    unsigned ldsK = (unsigned)(uintptr_t)&sK[0][0];
#endif

    // Q fragments: contraction over HD=64 -> two K=32 steps, loaded once
    Frag aq[2];
    {
        const bf16* qp = Q + (size_t)(b * SEQ + q0 + ml) * D_MODEL + h * HD;
#pragma unroll
        for (int ks = 0; ks < 2; ks++) {
            aq[ks].u[0] = *(const uint4*)(qp + ks * 32 + kh * 16);
            aq[ks].u[1] = *(const uint4*)(qp + ks * 32 + kh * 16 + 8);
        }
    }

    f32x8 o[4];
#pragma unroll
    for (int j = 0; j < 4; j++) o[j] = zero8();
    float rowmax[8], rowsum[8];
#pragma unroll
    for (int r = 0; r < 8; r++) { rowmax[r] = -3.0e38f; rowsum[r] = 0.0f; }

    int nkt = qt + 1;  // causal: key tiles 0..qt
    for (int kt = 0; kt < nkt; kt++) {
        int k0g = kt * 64;
        __syncthreads();  // protect LDS reuse from previous iteration

#if HAVE_TDM
        // --- TDM: DMA the 64x64 bf16 K tile into sK, padding rows 64->72 elems
        if (wave == 0) {
            uint64_t ga = (uint64_t)(uintptr_t)(Kt + (size_t)(b * SEQ + k0g) * D_MODEL + h * HD);
            u32x4 g0;
            g0[0] = 1u;                                                // count=1 (valid)
            g0[1] = ldsK;                                              // lds_addr
            g0[2] = (unsigned)ga;                                      // global_addr[31:0]
            g0[3] = (unsigned)((ga >> 32) & 0x01FFFFFFu) | 0x80000000u; // addr[56:32] | type=2
            i32x8 g1;
            g1[0] = (1 << 16)          // data_size = 2 bytes
                  | (1 << 20)          // pad_enable
                  | (4 << 22)          // pad_interval: 32 DWORDs (=one 128B row)
                  | (3 << 25);         // pad_amount:   4 DWORDs  (=16B -> 72-elem rows)
            g1[1] = 64 << 16;          // tensor_dim0[15:0] = 64
            g1[2] = 64 << 16;          // tensor_dim0[31:16]=0 | tensor_dim1[15:0] = 64
            g1[3] = 64 << 16;          // tensor_dim1[31:16]=0 | tile_dim0 = 64
            g1[4] = 64;                // tile_dim1 = 64, tile_dim2 = 0
            g1[5] = 512;               // tensor_dim0_stride[31:0] = D_MODEL
            g1[6] = 0;                 // stride0[47:32]=0 | tensor_dim1_stride lo = 0
            g1[7] = 0;
            i32x4 z4 = {0, 0, 0, 0};
#if __has_include(<hip/amd_detail/amd_gfx1250_TDM.h>)
            i32x8 z8 = {0, 0, 0, 0, 0, 0, 0, 0};
            __builtin_amdgcn_tensor_load_to_lds(g0, g1, z4, z4, z8, 0);
#else
            __builtin_amdgcn_tensor_load_to_lds(g0, g1, z4, z4, 0);
#endif
        }
        // --- overlap: all 128 threads stage V transposed
        for (int c = threadIdx.x; c < 512; c += 128) {
            int key = c >> 3;
            int dc  = (c & 7) * 8;
            size_t base = (size_t)(b * SEQ + k0g + key) * D_MODEL + h * HD + dc;
            V8 vv;
            vv.u = *(const uint4*)(Vt + base);
#pragma unroll
            for (int i = 0; i < 8; i++) sVT[dc + i][key] = vv.e[i];
        }
        if (wave == 0) __builtin_amdgcn_s_wait_tensorcnt(0);
        __syncthreads();
#else
        // fallback: cooperative stage of both K (row-major) and V (transposed)
        for (int c = threadIdx.x; c < 512; c += 128) {
            int key = c >> 3;
            int dc  = (c & 7) * 8;
            size_t base = (size_t)(b * SEQ + k0g + key) * D_MODEL + h * HD + dc;
            *(uint4*)&sK[key][dc] = *(const uint4*)(Kt + base);
            V8 vv;
            vv.u = *(const uint4*)(Vt + base);
#pragma unroll
            for (int i = 0; i < 8; i++) sVT[dc + i][key] = vv.e[i];
        }
        __syncthreads();
#endif

        // scores for 4 groups of 16 keys (each: 2 chained WMMA over hd)
        f32x8 st[4];
#pragma unroll
        for (int g = 0; g < 4; g++) {
            Frag b0, b1;
            int keyl = g * 16 + ml;
            b0.u[0] = *(const uint4*)&sK[keyl][kh * 16];
            b0.u[1] = *(const uint4*)&sK[keyl][kh * 16 + 8];
            b1.u[0] = *(const uint4*)&sK[keyl][32 + kh * 16];
            b1.u[1] = *(const uint4*)&sK[keyl][32 + kh * 16 + 8];
            f32x8 s = zero8();
            s = wmma_bf16(aq[0], b0, s);
            s = wmma_bf16(aq[1], b1, s);
            int kg = k0g + keyl;
#pragma unroll
            for (int r = 0; r < 8; r++) {
                int qr = q0 + kh * 8 + r;
                st[g][r] = (kg <= qr) ? s[r] * scale : -3.0e38f;
            }
        }

        // online softmax (row lives across 16 lanes of a half-wave)
#pragma unroll
        for (int r = 0; r < 8; r++) {
            float mx = fmaxf(fmaxf(st[0][r], st[1][r]), fmaxf(st[2][r], st[3][r]));
#pragma unroll
            for (int off = 8; off >= 1; off >>= 1) mx = fmaxf(mx, __shfl_xor(mx, off, 32));
            float nm = fmaxf(rowmax[r], mx);
            float sf = __expf(rowmax[r] - nm);
            rowmax[r] = nm;
            float ss = 0.0f;
#pragma unroll
            for (int g = 0; g < 4; g++) {
                float p = __expf(st[g][r] - nm);
                st[g][r] = p;
                ss += p;
            }
#pragma unroll
            for (int off = 8; off >= 1; off >>= 1) ss += __shfl_xor(ss, off, 32);
            rowsum[r] = rowsum[r] * sf + ss;
#pragma unroll
            for (int j = 0; j < 4; j++) o[j][r] *= sf;
        }

        // repack probs via LDS into A-fragment layout
#pragma unroll
        for (int g = 0; g < 4; g++)
#pragma unroll
            for (int r = 0; r < 8; r++)
                sP[wave][kh * 8 + r][g * 16 + ml] = (bf16)st[g][r];
        asm volatile("s_wait_dscnt 0" ::: "memory");

        // P @ V (contraction over 64 keys -> 2 chunks of 32)
#pragma unroll
        for (int kc = 0; kc < 2; kc++) {
            Frag ap;
            ap.u[0] = *(const uint4*)&sP[wave][ml][kc * 32 + kh * 16];
            ap.u[1] = *(const uint4*)&sP[wave][ml][kc * 32 + kh * 16 + 8];
#pragma unroll
            for (int j = 0; j < 4; j++) {
                Frag bv;
                bv.u[0] = *(const uint4*)&sVT[j * 16 + ml][kc * 32 + kh * 16];
                bv.u[1] = *(const uint4*)&sVT[j * 16 + ml][kc * 32 + kh * 16 + 8];
                o[j] = wmma_bf16(ap, bv, o[j]);
            }
        }
    }

    // epilogue: normalize, store context (bf16)
#pragma unroll
    for (int j = 0; j < 4; j++)
#pragma unroll
        for (int r = 0; r < 8; r++) {
            int qr = q0 + kh * 8 + r;
            float val = o[j][r] / rowsum[r];
            Ctx[(size_t)(b * SEQ + qr) * D_MODEL + h * HD + j * 16 + ml] = (bf16)val;
        }
}

// ---------------------------------------------------------------------------
// residual add + LayerNorm; one wave per row of 512
// ---------------------------------------------------------------------------
__global__ __launch_bounds__(256) void k_res_ln(float* __restrict__ x,
                                                const float* __restrict__ a,
                                                const float* __restrict__ g,
                                                const float* __restrict__ bta,
                                                bf16* __restrict__ xb) {
    int wid  = (blockIdx.x * 256 + threadIdx.x) >> 5;
    int lane = threadIdx.x & 31;
    if (wid >= NROWS) return;
    const float* xr = x + (size_t)wid * D_MODEL;
    const float* ar = a + (size_t)wid * D_MODEL;
    float v[16];
    float s = 0.0f;
#pragma unroll
    for (int i = 0; i < 16; i++) {
        v[i] = xr[lane + i * 32] + ar[lane + i * 32];
        s += v[i];
    }
#pragma unroll
    for (int off = 16; off >= 1; off >>= 1) s += __shfl_xor(s, off, 32);
    float mu = s * (1.0f / 512.0f);
    float vs = 0.0f;
#pragma unroll
    for (int i = 0; i < 16; i++) { float d = v[i] - mu; vs += d * d; }
#pragma unroll
    for (int off = 16; off >= 1; off >>= 1) vs += __shfl_xor(vs, off, 32);
    float rstd = rsqrtf(vs * (1.0f / 512.0f) + LN_EPS);
#pragma unroll
    for (int i = 0; i < 16; i++) {
        int cidx = lane + i * 32;
        float y = (v[i] - mu) * rstd * g[cidx] + bta[cidx];
        x[(size_t)wid * D_MODEL + cidx]  = y;
        xb[(size_t)wid * D_MODEL + cidx] = (bf16)y;
    }
}

// ---------------------------------------------------------------------------
// final projection + tanh; one wave per row
// ---------------------------------------------------------------------------
__global__ __launch_bounds__(256) void k_final(const float* __restrict__ x,
                                               const float* __restrict__ w,
                                               const float* __restrict__ b,
                                               float* __restrict__ out) {
    int wid  = (blockIdx.x * 256 + threadIdx.x) >> 5;
    int lane = threadIdx.x & 31;
    if (wid >= NROWS) return;
    const float* xr = x + (size_t)wid * D_MODEL;
    float s = 0.0f;
#pragma unroll
    for (int i = 0; i < 16; i++) s += xr[lane + i * 32] * w[lane + i * 32];
#pragma unroll
    for (int off = 16; off >= 1; off >>= 1) s += __shfl_xor(s, off, 32);
    if (lane == 0) out[wid] = 1.5f * tanhf(s + b[0]);
}

// ---------------------------------------------------------------------------
extern "C" void kernel_launch(void* const* d_in, const int* in_sizes, int n_in,
                              void* d_out, int out_size, void* d_ws, size_t ws_size,
                              hipStream_t stream) {
    const float* feat       = (const float*)d_in[0];
    const float* hist_w     = (const float*)d_in[1];
    const float* hist_b     = (const float*)d_in[2];
    const float* query_w    = (const float*)d_in[3];
    const float* query_b    = (const float*)d_in[4];
    const float* in_proj_w  = (const float*)d_in[5];
    const float* in_proj_b  = (const float*)d_in[6];
    const float* attn_out_w = (const float*)d_in[7];
    const float* attn_out_b = (const float*)d_in[8];
    const float* ln_g       = (const float*)d_in[9];
    const float* ln_b       = (const float*)d_in[10];
    const float* out_w      = (const float*)d_in[11];
    const float* out_b      = (const float*)d_in[12];
    float* out = (float*)d_out;

    char* ws = (char*)d_ws;
    size_t off = 0;
    auto alloc = [&](size_t bytes) -> void* {
        void* p = ws + off;
        off += (bytes + 255) & ~(size_t)255;
        return p;
    };
    float* x_f32  = (float*)alloc((size_t)NROWS * D_MODEL * 4);
    float* a_f32  = (float*)alloc((size_t)NROWS * D_MODEL * 4);
    bf16*  x_bf   = (bf16*)alloc((size_t)NROWS * D_MODEL * 2);
    bf16*  h_bf   = (bf16*)alloc((size_t)NROWS * D_MODEL * 2);
    bf16*  q_bf   = (bf16*)alloc((size_t)NROWS * D_MODEL * 2);
    bf16*  k_bf   = (bf16*)alloc((size_t)NROWS * D_MODEL * 2);
    bf16*  v_bf   = (bf16*)alloc((size_t)NROWS * D_MODEL * 2);
    bf16*  ctx_bf = (bf16*)alloc((size_t)NROWS * D_MODEL * 2);
    bf16*  wqkv   = (bf16*)alloc((size_t)2 * 1536 * 512 * 2);
    bf16*  wo     = (bf16*)alloc((size_t)2 * 512 * 512 * 2);

    // weight conversion
    k_f32_to_bf16<<<(2 * 1536 * 512) / 256, 256, 0, stream>>>(in_proj_w, wqkv, 2 * 1536 * 512);
    k_f32_to_bf16<<<(2 * 512 * 512) / 256, 256, 0, stream>>>(attn_out_w, wo, 2 * 512 * 512);

    // input projections
    k_input_proj<<<(NROWS * D_MODEL) / 256, 256, 0, stream>>>(
        feat, hist_w, hist_b, query_w, query_b, x_f32, x_bf, h_bf);

    const int gemmBlocks = (NROWS / 32) * (512 / 64) / 8;  // 512

    for (int l = 0; l < 2; l++) {
        const bf16*  wl = wqkv + (size_t)l * 1536 * 512;
        const float* bl = in_proj_b + l * 1536;
        // Q from x, K/V from history
        k_gemm<<<gemmBlocks, 256, 0, stream>>>(x_bf, wl,              bl,        nullptr, q_bf, NROWS, 512, 512);
        k_gemm<<<gemmBlocks, 256, 0, stream>>>(h_bf, wl + 512 * 512,  bl + 512,  nullptr, k_bf, NROWS, 512, 512);
        k_gemm<<<gemmBlocks, 256, 0, stream>>>(h_bf, wl + 1024 * 512, bl + 1024, nullptr, v_bf, NROWS, 512, 512);
        // attention
        k_attn<<<BATCH * NHEAD * (SEQ / 64), 128, 0, stream>>>(q_bf, k_bf, v_bf, ctx_bf);
        // output projection (fp32 out for residual/LN)
        k_gemm<<<gemmBlocks, 256, 0, stream>>>(ctx_bf, wo + (size_t)l * 512 * 512,
                                               attn_out_b + l * 512, a_f32, nullptr, NROWS, 512, 512);
        // residual + LayerNorm
        k_res_ln<<<NROWS / 8, 256, 0, stream>>>(x_f32, a_f32, ln_g + l * 512, ln_b + l * 512, x_bf);
    }

    k_final<<<NROWS / 8, 256, 0, stream>>>(x_f32, out_w, out_b, out);
}
